// GATCF2_82858509074813
// MI455X (gfx1250) — matrix operand order repaired and checked by reading, not compiled, not run on hardware
//
#include <hip/hip_runtime.h>
#include <hip/hip_bf16.h>
#include <math.h>

#define NHEAD 2
#define DIMF  64
#define HDF   128

typedef __attribute__((ext_vector_type(16))) _Float16 v16h;
typedef __attribute__((ext_vector_type(8)))  _Float16 v8h;
typedef __attribute__((ext_vector_type(8)))  float    v8f;

// ---------------------------------------------------------------- utilities
__device__ __forceinline__ unsigned fkey(float f) {
  unsigned b = __float_as_uint(f);
  return (b & 0x80000000u) ? ~b : (b | 0x80000000u);  // order-preserving key
}
__device__ __forceinline__ float funkey(unsigned k) {
  unsigned b = (k & 0x80000000u) ? (k & 0x7FFFFFFFu) : ~k;
  return __uint_as_float(b);
}

__global__ __launch_bounds__(256) void k_zero_u32(unsigned* __restrict__ p, size_t n) {
  size_t i = (size_t)blockIdx.x * blockDim.x + threadIdx.x;
  if (i < n) p[i] = 0u;
}

__global__ __launch_bounds__(256) void k_f32_to_f16(const float* __restrict__ in,
                                                    _Float16* __restrict__ out, size_t n) {
  size_t i = (size_t)blockIdx.x * blockDim.x + threadIdx.x;
  if (i < n) out[i] = (_Float16)in[i];
}

// in: [K x N] f32 row-major  ->  out: [N x K] f16 (i.e. B transposed, col-contiguous)
__global__ __launch_bounds__(256)
void k_transpose_f32_to_f16(const float* __restrict__ in, _Float16* __restrict__ out,
                            int K, int N) {
  int i = blockIdx.x * blockDim.x + threadIdx.x;
  if (i >= K * N) return;
  int k = i / N, n = i - k * N;
  out[(size_t)n * K + k] = (_Float16)in[i];
}

// ---------------------------------------------------------------- WMMA GEMM
// C[M x N] (f32) = A[M x K] (f16 row major) * B[K x N], with B given as BT[N x K] f16.
// Block = 128 threads = 4 waves; each wave computes one 16-row stripe.
// BT is staged in LDS (padded pitch); B fragments are double-buffered so LDS
// reads for tile t+1 are in flight while tile t's v_wmma issue.
// Lane layouts per CDNA5 ISA 7.12.2:
//   A 16x32 f16 : lanes 0-15 rows M=0..15, K = 8*(lane>=16) + (i<8 ? i : i+8)
//   B 32x16 f16 : lanes 0-15 cols N=0..15, K = 8*(lane>=16) + (i<8 ? i : i+8)
//   C 16x16 f32 : VGPR r -> M = r + 8*(lane>=16), N = lane&15
template <int K, int N>
__global__ __launch_bounds__(128)
void k_gemm_f16_wmma(const _Float16* __restrict__ A, const _Float16* __restrict__ BT,
                     float* __restrict__ C, int M) {
  constexpr int NK    = K / 32;
  constexpr int NT    = N / 16;
  constexpr int PITCH = K + 8;                       // halves; pad to spread LDS banks
  __shared__ _Float16 bsh[N * PITCH];

  const int tid = threadIdx.x;
  // cooperative stage of BT into LDS (vectorized 8-half chunks)
  for (int idx = tid; idx < N * (K / 8); idx += 128) {
    int col = idx / (K / 8);
    int kk  = (idx - col * (K / 8)) * 8;
    *(v8h*)&bsh[col * PITCH + kk] = *(const v8h*)&BT[(size_t)col * K + kk];
  }
  __syncthreads();

  const int wave     = tid >> 5;
  const int lane     = tid & 31;
  const int half_sel = lane >> 4;
  const int lane16   = lane & 15;
  const int m0       = (blockIdx.x * 4 + wave) * 16;
  if (m0 >= M) return;                               // uniform whole-wave skip

  // ---- A fragments: two 16B global loads each
  v16h afrag[NK];
  {
    int row = m0 + lane16;
    if (row >= M) row = M - 1;                       // tail clamp (stores guarded below)
    const _Float16* abase = A + (size_t)row * K + half_sel * 8;
    #pragma unroll
    for (int kc = 0; kc < NK; ++kc) {
      v8h lo = *(const v8h*)(abase + kc * 32);
      v8h hi = *(const v8h*)(abase + kc * 32 + 16);
      afrag[kc] = __builtin_shufflevector(lo, hi, 0, 1, 2, 3, 4, 5, 6, 7,
                                          8, 9, 10, 11, 12, 13, 14, 15);
    }
  }

  const bool full = (m0 + 16 <= M);
  const int  bcol = lane16 * PITCH + half_sel * 8;   // halves offset into bsh row

  auto loadB = [&](int t, int kc) -> v16h {
    const _Float16* bb = &bsh[(t * 16) * PITCH + bcol + kc * 32];
    v8h lo = *(const v8h*)(bb);
    v8h hi = *(const v8h*)(bb + 16);
    return __builtin_shufflevector(lo, hi, 0, 1, 2, 3, 4, 5, 6, 7,
                                   8, 9, 10, 11, 12, 13, 14, 15);
  };

  v8f acc[NT];
  #pragma unroll
  for (int t = 0; t < NT; ++t) acc[t] = (v8f){};

  // ---- double-buffered B fragments across column tiles
  v16h bcur[NK], bnxt[NK];
  #pragma unroll
  for (int kc = 0; kc < NK; ++kc) bcur[kc] = loadB(0, kc);

  #pragma unroll
  for (int t = 0; t < NT; ++t) {
    if (t + 1 < NT) {
      #pragma unroll
      for (int kc = 0; kc < NK; ++kc) bnxt[kc] = loadB(t + 1, kc);
    }
    #pragma unroll
    for (int kc = 0; kc < NK; ++kc)
      acc[t] = __builtin_amdgcn_wmma_f32_16x16x32_f16(
          false, afrag[kc], false, bcur[kc], (short)0, acc[t], false, false);
    if (t + 1 < NT) {
      #pragma unroll
      for (int kc = 0; kc < NK; ++kc) bcur[kc] = bnxt[kc];
    }
  }

  const int rbase = half_sel * 8;
  if (full) {
    #pragma unroll
    for (int t = 0; t < NT; ++t) {
      float* crow = C + (size_t)(m0 + rbase) * N + t * 16 + lane16;
      #pragma unroll
      for (int r = 0; r < 8; ++r) crow[(size_t)r * N] = acc[t][r];
    }
  } else {
    #pragma unroll
    for (int t = 0; t < NT; ++t)
      #pragma unroll
      for (int r = 0; r < 8; ++r) {
        int row = m0 + rbase + r;
        if (row < M) C[(size_t)row * N + t * 16 + lane16] = acc[t][r];
      }
  }
}

// ---------------------------------------------------------------- GAT pieces
__global__ __launch_bounds__(256)
void k_el_er(const float* __restrict__ h, const float* __restrict__ al,
             const float* __restrict__ ar, float* __restrict__ el,
             float* __restrict__ er, int n) {
  int i = blockIdx.x * blockDim.x + threadIdx.x;
  if (i >= n * NHEAD) return;
  int node = i >> 1, hh = i & 1;
  const float* hp  = h  + (size_t)node * HDF + hh * DIMF;
  const float* alp = al + hh * DIMF;
  const float* arp = ar + hh * DIMF;
  float sl = 0.f, sr = 0.f;
  #pragma unroll 4
  for (int d = 0; d < DIMF; ++d) { float v = hp[d]; sl += v * alp[d]; sr += v * arp[d]; }
  el[i] = sl; er[i] = sr;
}

__global__ __launch_bounds__(256)
void k_edge_max(const int* __restrict__ src, const int* __restrict__ dst,
                const float* __restrict__ el, const float* __restrict__ er,
                unsigned* __restrict__ mkey, int E) {
  int e = blockIdx.x * blockDim.x + threadIdx.x;
  if (e >= E) return;
  int s = src[e], d = dst[e];
  #pragma unroll
  for (int hh = 0; hh < NHEAD; ++hh) {
    float x = el[s * NHEAD + hh] + er[d * NHEAD + hh];
    x = (x > 0.f) ? x : 0.2f * x;           // leaky_relu 0.2
    atomicMax(&mkey[d * NHEAD + hh], fkey(x));
  }
}

__global__ __launch_bounds__(256)
void k_edge_expsum(const int* __restrict__ src, const int* __restrict__ dst,
                   const float* __restrict__ el, const float* __restrict__ er,
                   const unsigned* __restrict__ mkey, float* __restrict__ den, int E) {
  int e = blockIdx.x * blockDim.x + threadIdx.x;
  if (e >= E) return;
  int s = src[e], d = dst[e];
  #pragma unroll
  for (int hh = 0; hh < NHEAD; ++hh) {
    float x = el[s * NHEAD + hh] + er[d * NHEAD + hh];
    x = (x > 0.f) ? x : 0.2f * x;
    float ex = __expf(x - funkey(mkey[d * NHEAD + hh]));
    atomicAdd(&den[d * NHEAD + hh], ex);
  }
}

__global__ __launch_bounds__(128)
void k_edge_accum(const int* __restrict__ src, const int* __restrict__ dst,
                  const float* __restrict__ el, const float* __restrict__ er,
                  const unsigned* __restrict__ mkey, const float* __restrict__ den,
                  const float* __restrict__ h, float* __restrict__ acc) {
  int e  = blockIdx.x;
  int j  = threadIdx.x;       // 0..127 -> (head, dim)
  int hh = j >> 6;
  int s = src[e], d = dst[e];
  float x = el[s * NHEAD + hh] + er[d * NHEAD + hh];
  x = (x > 0.f) ? x : 0.2f * x;
  float alpha = __expf(x - funkey(mkey[d * NHEAD + hh])) / den[d * NHEAD + hh];
  atomicAdd(&acc[(size_t)d * HDF + j], alpha * h[(size_t)s * HDF + j]);
}

// bias + LayerNorm(128) + ELU + mean over 2 heads -> [N,64]
__global__ __launch_bounds__(128)
void k_gat_finalize(const float* __restrict__ acc, const float* __restrict__ bias,
                    const float* __restrict__ lng, const float* __restrict__ lnb,
                    float* __restrict__ outg) {
  __shared__ float vals[128];
  __shared__ float rbuf[128];
  int node = blockIdx.x, j = threadIdx.x;
  float v = acc[(size_t)node * HDF + j] + bias[j];
  rbuf[j] = v; __syncthreads();
  for (int s = 64; s; s >>= 1) { if (j < s) rbuf[j] += rbuf[j + s]; __syncthreads(); }
  float mu = rbuf[0] * (1.0f / 128.0f); __syncthreads();
  float dd = v - mu;
  rbuf[j] = dd * dd; __syncthreads();
  for (int s = 64; s; s >>= 1) { if (j < s) rbuf[j] += rbuf[j + s]; __syncthreads(); }
  float rs = rsqrtf(rbuf[0] * (1.0f / 128.0f) + 1e-5f);
  float y  = dd * rs * lng[j] + lnb[j];
  y = (y > 0.f) ? y : (__expf(y) - 1.0f);   // ELU
  vals[j] = y; __syncthreads();
  if (j < DIMF) outg[(size_t)node * DIMF + j] = 0.5f * (vals[j] + vals[j + DIMF]);
}

// ---------------------------------------------------------------- MLP pieces
__global__ __launch_bounds__(128)
void k_gather_concat(const int* __restrict__ uIdx, const int* __restrict__ sIdx,
                     const float* __restrict__ gu, const float* __restrict__ gs,
                     const float* __restrict__ ue, const float* __restrict__ se,
                     _Float16* __restrict__ x16) {
  int b = blockIdx.x, j = threadIdx.x;
  float v;
  if (j < DIMF) {
    int u = uIdx[b];
    v = gu[(size_t)u * DIMF + j] + ue[(size_t)u * DIMF + j];
  } else {
    int s = sIdx[b]; int d = j - DIMF;
    v = gs[(size_t)s * DIMF + d] + se[(size_t)s * DIMF + d];
  }
  x16[(size_t)b * HDF + j] = (_Float16)v;
}

// (y + b) -> LayerNorm(128) -> ReLU ; writes f32 and optional f16
__global__ __launch_bounds__(128)
void k_ln_relu(const float* __restrict__ y, const float* __restrict__ bv,
               const float* __restrict__ lng, const float* __restrict__ lnb,
               float* __restrict__ of, _Float16* __restrict__ oh) {
  __shared__ float rbuf[128];
  int j = threadIdx.x;
  size_t base = (size_t)blockIdx.x * HDF;
  float v = y[base + j] + bv[j];
  rbuf[j] = v; __syncthreads();
  for (int s = 64; s; s >>= 1) { if (j < s) rbuf[j] += rbuf[j + s]; __syncthreads(); }
  float mu = rbuf[0] * (1.0f / 128.0f); __syncthreads();
  float dd = v - mu;
  rbuf[j] = dd * dd; __syncthreads();
  for (int s = 64; s; s >>= 1) { if (j < s) rbuf[j] += rbuf[j + s]; __syncthreads(); }
  float rs = rsqrtf(rbuf[0] * (1.0f / 128.0f) + 1e-5f);
  float o  = fmaxf(dd * rs * lng[j] + lnb[j], 0.0f);
  of[base + j] = o;
  if (oh) oh[base + j] = (_Float16)o;
}

__global__ __launch_bounds__(256)
void k_final_dot_sigmoid(const float* __restrict__ x, const float* __restrict__ W3,
                         const float* __restrict__ b3, float* __restrict__ out, int B) {
  int b = blockIdx.x * blockDim.x + threadIdx.x;
  if (b >= B) return;
  const float* xp = x + (size_t)b * HDF;
  float sum = b3[0];
  #pragma unroll 4
  for (int j = 0; j < HDF; ++j) sum += xp[j] * W3[j];
  out[b] = 1.0f / (1.0f + __expf(-sum));
}

// ---------------------------------------------------------------- host side
static inline size_t align256(size_t x) { return (x + 255) & ~(size_t)255; }

extern "C" void kernel_launch(void* const* d_in, const int* in_sizes, int n_in,
                              void* d_out, int out_size, void* d_ws, size_t ws_size,
                              hipStream_t stream) {
  // input order per setup_inputs()
  const int*   userIdx  = (const int*)  d_in[0];
  const int*   servIdx  = (const int*)  d_in[1];
  const int*   u_src    = (const int*)  d_in[2];
  const int*   u_dst    = (const int*)  d_in[3];
  const int*   s_src    = (const int*)  d_in[4];
  const int*   s_dst    = (const int*)  d_in[5];
  const float* user_emb = (const float*)d_in[6];
  const float* serv_emb = (const float*)d_in[7];
  const float* u_W      = (const float*)d_in[8];
  const float* u_al     = (const float*)d_in[9];
  const float* u_ar     = (const float*)d_in[10];
  const float* u_bias   = (const float*)d_in[11];
  const float* u_lng    = (const float*)d_in[12];
  const float* u_lnb    = (const float*)d_in[13];
  const float* s_W      = (const float*)d_in[14];
  const float* s_al     = (const float*)d_in[15];
  const float* s_ar     = (const float*)d_in[16];
  const float* s_bias   = (const float*)d_in[17];
  const float* s_lng    = (const float*)d_in[18];
  const float* s_lnb    = (const float*)d_in[19];
  const float* W1       = (const float*)d_in[20];
  const float* b1       = (const float*)d_in[21];
  const float* ln1_g    = (const float*)d_in[22];
  const float* ln1_b    = (const float*)d_in[23];
  const float* W2       = (const float*)d_in[24];
  const float* b2       = (const float*)d_in[25];
  const float* ln2_g    = (const float*)d_in[26];
  const float* ln2_b    = (const float*)d_in[27];
  const float* W3       = (const float*)d_in[28];
  const float* b3       = (const float*)d_in[29];

  const int B  = in_sizes[0];
  const int EU = in_sizes[2];
  const int ES = in_sizes[4];
  const int NU = in_sizes[6] / DIMF;
  const int NS = in_sizes[7] / DIMF;

  // bump allocator over d_ws
  char* ws = (char*)d_ws;
  size_t off = 0;
  auto alloc = [&](size_t bytes) -> void* {
    void* p = ws + off;
    off += align256(bytes);
    return p;
  };

  _Float16* uemb16 = (_Float16*)alloc((size_t)NU * DIMF * 2);
  _Float16* semb16 = (_Float16*)alloc((size_t)NS * DIMF * 2);
  _Float16* uWT16  = (_Float16*)alloc((size_t)DIMF * HDF * 2);  // [128 x 64]
  _Float16* sWT16  = (_Float16*)alloc((size_t)DIMF * HDF * 2);
  _Float16* W1T16  = (_Float16*)alloc((size_t)HDF * HDF * 2);   // [128 x 128]
  _Float16* W2T16  = (_Float16*)alloc((size_t)HDF * HDF * 2);
  float*    h_u    = (float*)   alloc((size_t)NU * HDF * 4);
  float*    h_s    = (float*)   alloc((size_t)NS * HDF * 4);
  float*    el_u   = (float*)   alloc((size_t)NU * NHEAD * 4);
  float*    er_u   = (float*)   alloc((size_t)NU * NHEAD * 4);
  float*    el_s   = (float*)   alloc((size_t)NS * NHEAD * 4);
  float*    er_s   = (float*)   alloc((size_t)NS * NHEAD * 4);
  unsigned* mk_u   = (unsigned*)alloc((size_t)NU * NHEAD * 4);
  unsigned* mk_s   = (unsigned*)alloc((size_t)NS * NHEAD * 4);
  float*    den_u  = (float*)   alloc((size_t)NU * NHEAD * 4);
  float*    den_s  = (float*)   alloc((size_t)NS * NHEAD * 4);
  float*    acc_u  = (float*)   alloc((size_t)NU * HDF * 4);
  float*    acc_s  = (float*)   alloc((size_t)NS * HDF * 4);
  float*    g_u    = (float*)   alloc((size_t)NU * DIMF * 4);
  float*    g_s    = (float*)   alloc((size_t)NS * DIMF * 4);
  _Float16* x16    = (_Float16*)alloc((size_t)B * HDF * 2);
  float*    y1     = (float*)   alloc((size_t)B * HDF * 4);
  float*    x1f    = (float*)   alloc((size_t)B * HDF * 4);
  _Float16* x1h    = (_Float16*)alloc((size_t)B * HDF * 2);
  float*    y2     = (float*)   alloc((size_t)B * HDF * 4);
  float*    x2f    = (float*)   alloc((size_t)B * HDF * 4);
  (void)ws_size; (void)n_in; (void)out_size;

  auto cdiv = [](size_t a, size_t b) { return (unsigned)((a + b - 1) / b); };

  // ---- f16 conversions / weight transposes
  k_f32_to_f16<<<cdiv((size_t)NU * DIMF, 256), 256, 0, stream>>>(user_emb, uemb16, (size_t)NU * DIMF);
  k_f32_to_f16<<<cdiv((size_t)NS * DIMF, 256), 256, 0, stream>>>(serv_emb, semb16, (size_t)NS * DIMF);
  k_transpose_f32_to_f16<<<cdiv((size_t)DIMF * HDF, 256), 256, 0, stream>>>(u_W, uWT16, DIMF, HDF);
  k_transpose_f32_to_f16<<<cdiv((size_t)DIMF * HDF, 256), 256, 0, stream>>>(s_W, sWT16, DIMF, HDF);
  k_transpose_f32_to_f16<<<cdiv((size_t)HDF * HDF, 256), 256, 0, stream>>>(W1, W1T16, HDF, HDF);
  k_transpose_f32_to_f16<<<cdiv((size_t)HDF * HDF, 256), 256, 0, stream>>>(W2, W2T16, HDF, HDF);

  // ---- GAT: user graph
  k_gemm_f16_wmma<64, 128><<<cdiv(NU, 64), 128, 0, stream>>>(uemb16, uWT16, h_u, NU);
  k_el_er<<<cdiv((size_t)NU * NHEAD, 256), 256, 0, stream>>>(h_u, u_al, u_ar, el_u, er_u, NU);
  k_zero_u32<<<cdiv((size_t)NU * NHEAD, 256), 256, 0, stream>>>(mk_u, (size_t)NU * NHEAD);
  k_zero_u32<<<cdiv((size_t)NU * NHEAD, 256), 256, 0, stream>>>((unsigned*)den_u, (size_t)NU * NHEAD);
  k_zero_u32<<<cdiv((size_t)NU * HDF, 256), 256, 0, stream>>>((unsigned*)acc_u, (size_t)NU * HDF);
  k_edge_max<<<cdiv(EU, 256), 256, 0, stream>>>(u_src, u_dst, el_u, er_u, mk_u, EU);
  k_edge_expsum<<<cdiv(EU, 256), 256, 0, stream>>>(u_src, u_dst, el_u, er_u, mk_u, den_u, EU);
  k_edge_accum<<<EU, 128, 0, stream>>>(u_src, u_dst, el_u, er_u, mk_u, den_u, h_u, acc_u);
  k_gat_finalize<<<NU, 128, 0, stream>>>(acc_u, u_bias, u_lng, u_lnb, g_u);

  // ---- GAT: service graph
  k_gemm_f16_wmma<64, 128><<<cdiv(NS, 64), 128, 0, stream>>>(semb16, sWT16, h_s, NS);
  k_el_er<<<cdiv((size_t)NS * NHEAD, 256), 256, 0, stream>>>(h_s, s_al, s_ar, el_s, er_s, NS);
  k_zero_u32<<<cdiv((size_t)NS * NHEAD, 256), 256, 0, stream>>>(mk_s, (size_t)NS * NHEAD);
  k_zero_u32<<<cdiv((size_t)NS * NHEAD, 256), 256, 0, stream>>>((unsigned*)den_s, (size_t)NS * NHEAD);
  k_zero_u32<<<cdiv((size_t)NS * HDF, 256), 256, 0, stream>>>((unsigned*)acc_s, (size_t)NS * HDF);
  k_edge_max<<<cdiv(ES, 256), 256, 0, stream>>>(s_src, s_dst, el_s, er_s, mk_s, ES);
  k_edge_expsum<<<cdiv(ES, 256), 256, 0, stream>>>(s_src, s_dst, el_s, er_s, mk_s, den_s, ES);
  k_edge_accum<<<ES, 128, 0, stream>>>(s_src, s_dst, el_s, er_s, mk_s, den_s, h_s, acc_s);
  k_gat_finalize<<<NS, 128, 0, stream>>>(acc_s, s_bias, s_lng, s_lnb, g_s);

  // ---- MLP head
  k_gather_concat<<<B, 128, 0, stream>>>(userIdx, servIdx, g_u, g_s, user_emb, serv_emb, x16);
  k_gemm_f16_wmma<128, 128><<<cdiv(B, 64), 128, 0, stream>>>(x16, W1T16, y1, B);
  k_ln_relu<<<B, 128, 0, stream>>>(y1, b1, ln1_g, ln1_b, x1f, x1h);
  k_gemm_f16_wmma<128, 128><<<cdiv(B, 64), 128, 0, stream>>>(x1h, W2T16, y2, B);
  k_ln_relu<<<B, 128, 0, stream>>>(y2, b2, ln2_g, ln2_b, x2f, (_Float16*)nullptr);
  k_final_dot_sigmoid<<<cdiv(B, 256), 256, 0, stream>>>(x2f, W3, b3, (float*)d_out, B);
}
